// GRbOL_86131274154488
// MI455X (gfx1250) — compile-verified, matrix operand order checked by reading, not run on hardware
//
#include <hip/hip_runtime.h>
#include <hip/hip_bf16.h>

typedef __attribute__((ext_vector_type(16))) _Float16 v16h;
typedef __attribute__((ext_vector_type(8)))  float    v8f;
typedef int async_v4i __attribute__((vector_size(16)));   // pointee type of async-LDS builtins

#define BATCH   65536
#define MM      16
#define DD      32
#define NEGF    (-1e9f)

#define K1_BLOCKS  2048
#define K1_THREADS 128              // 4 waves of 32
#define K1_WAVES   (K1_BLOCKS * 4)  // 8192
#define K1_ITERS   (BATCH / K1_WAVES) // 8, identical for every wave -> uniform __syncthreads

#define CHUNK   256
#define NCHUNK  (BATCH / CHUNK)     // 256

#if __has_builtin(__builtin_amdgcn_update_dpp)
#define HAVE_DPP 1
#endif
#if __has_builtin(__builtin_amdgcn_global_load_async_to_lds_b128)
#define HAVE_ASYNC_LDS 1
#endif

#if HAVE_DPP
// v += dpp_permuted(v); CTRL is an immediate DPP16 control.
template <int CTRL>
__device__ __forceinline__ float dpp_add(float v) {
  const int m = __builtin_amdgcn_update_dpp(0, __float_as_int(v), CTRL, 0xf, 0xf, true);
  return v + __int_as_float(m);
}
#endif

// Sum over each 16-lane half (result uniform within the half).
__device__ __forceinline__ float half_sum16(float v) {
#if HAVE_DPP
  v = dpp_add<0x0B1>(v);  // quad_perm [1,0,3,2]  : pair xor1
  v = dpp_add<0x04E>(v);  // quad_perm [2,3,0,1]  : pair xor2 (quad-uniform now)
  v = dpp_add<0x141>(v);  // row_half_mirror      : pairs quads within 8
  v = dpp_add<0x140>(v);  // row_mirror           : pairs 8-groups within 16
  return v;
#else
  v += __shfl_xor(v, 1, 32);
  v += __shfl_xor(v, 2, 32);
  v += __shfl_xor(v, 4, 32);
  v += __shfl_xor(v, 8, 32);
  return v;
#endif
}
__device__ __forceinline__ float wave_sum32(float v) {
  v = half_sum16(v);
  v += __shfl_xor(v, 16, 32);   // cross-half combine
  return v;
}
__device__ __forceinline__ float bcast_lane(float v, int srcLane) {
#if __has_builtin(__builtin_amdgcn_readlane)
  return __int_as_float(__builtin_amdgcn_readlane(__float_as_int(v), srcLane));
#else
  return __shfl(v, srcLane, 32);
#endif
}

// Stage the 32-float row U[urow] into LDS (two lanes with same nrow cooperate:
// lane (nrow, h) covers cols [8h, 8h+8) and [16+8h, 16+8h+8)).
__device__ __forceinline__ void stage_member_row(const float* U, int urow,
                                                 float* dstRow, int halfi)
{
#if HAVE_ASYNC_LDS
  async_v4i __attribute__((address_space(1)))* src =
      (async_v4i __attribute__((address_space(1)))*)(U + (size_t)urow * DD + 8 * halfi);
  async_v4i __attribute__((address_space(3)))* dst =
      (async_v4i __attribute__((address_space(3)))*)(dstRow + 8 * halfi);
  // dsaddr/memaddr share INST_OFFSET: col+4 -> +16B, col+16 -> +64B
  __builtin_amdgcn_global_load_async_to_lds_b128(src, dst, 0, 0);
  __builtin_amdgcn_global_load_async_to_lds_b128(src, dst, 16, 0);
  __builtin_amdgcn_global_load_async_to_lds_b128(src, dst, 64, 0);
  __builtin_amdgcn_global_load_async_to_lds_b128(src, dst, 80, 0);
#else
  const float4* up = (const float4*)(U + (size_t)urow * DD);
  *(float4*)(dstRow + 8 * halfi)          = up[2 * halfi];
  *(float4*)(dstRow + 8 * halfi + 4)      = up[2 * halfi + 1];
  *(float4*)(dstRow + 16 + 8 * halfi)     = up[4 + 2 * halfi];
  *(float4*)(dstRow + 16 + 8 * halfi + 4) = up[5 + 2 * halfi];
#endif
}

template <int N>
__device__ __forceinline__ void wait_async() {
#if HAVE_ASYNC_LDS
#if __has_builtin(__builtin_amdgcn_s_wait_asynccnt)
  __builtin_amdgcn_s_wait_asynccnt(N);
#else
  asm volatile("s_wait_asynccnt %0" :: "i"(N) : "memory");
#endif
#endif
}

// ---------------------------------------------------------------------------
// Kernel 1: one wave per batch element. WMMA MLPs + g_ilm + sigmoid term.
// Member tiles are double-buffered: next iteration's async loads are issued
// before waiting on the current buffer (async loads complete in order).
// ---------------------------------------------------------------------------
__global__ void __launch_bounds__(K1_THREADS)
gilm_kernel(const int* __restrict__ group_in, const int* __restrict__ item_in,
            const int* __restrict__ members,  const int* __restrict__ lens,
            const float* __restrict__ U, const float* __restrict__ I,
            const float* __restrict__ G,
            const float* __restrict__ Wa1, const float* __restrict__ ba1,
            const float* __restrict__ Wa2,
            const float* __restrict__ Wu1, const float* __restrict__ bu1,
            const float* __restrict__ Wu2,
            const float* __restrict__ Wp1, const float* __restrict__ bp1,
            const float* __restrict__ Wp2, const float* __restrict__ bp2,
            float* __restrict__ gilm, float* __restrict__ sbuf)
{
  __shared__ _Float16 wa1h[64 * 16];      // Wa1 as f16, [k][n]
  __shared__ _Float16 wu1h[32 * 16];      // Wu1 as f16, [k][n]
  __shared__ float    wp1s[96 * 8];       // Wp1 f32, [k][h]
  __shared__ float    ba1s[16], wa2s[16], bu1s[16], wu2s[16], bp1s[8], wp2s[8];
  __shared__ float    memf[4][2][16][36]; // per-wave double-buffered member tile
  __shared__ float    itf[4][32];         // per-wave item embedding

  const int tid = threadIdx.x;
  for (int i = tid; i < 64 * 16; i += K1_THREADS) wa1h[i] = (_Float16)Wa1[i];
  for (int i = tid; i < 32 * 16; i += K1_THREADS) wu1h[i] = (_Float16)Wu1[i];
  for (int i = tid; i < 96 * 8;  i += K1_THREADS) wp1s[i] = Wp1[i];
  if (tid < 16) { ba1s[tid] = ba1[tid]; wa2s[tid] = Wa2[tid];
                  bu1s[tid] = bu1[tid]; wu2s[tid] = Wu2[tid]; }
  if (tid < 8)  { bp1s[tid] = bp1[tid]; wp2s[tid] = Wp2[tid]; }
  __syncthreads();

  const int lane  = tid & 31;
  const int w     = tid >> 5;
  const int halfi = lane >> 4;        // which 16-lane half
  const int nrow  = lane & 15;        // matrix row (A) / column (B)
  const int base  = 8 * halfi;        // K base for this half
  const float bp2v = bp2[0];

  // Build B-matrix fragments once (weights are loop-invariant).
  // 16-bit B 32x16: elem j -> W[base+j][n], elem 8+j -> W[16+base+j][n]
  v16h fwa0, fwa1, fwu;
#pragma unroll
  for (int j = 0; j < 8; ++j) {
    fwa0[j]     = wa1h[(base + j) * 16 + nrow];
    fwa0[8 + j] = wa1h[(16 + base + j) * 16 + nrow];
    fwa1[j]     = wa1h[(32 + base + j) * 16 + nrow];
    fwa1[8 + j] = wa1h[(48 + base + j) * 16 + nrow];
    fwu[j]      = wu1h[(base + j) * 16 + nrow];
    fwu[8 + j]  = wu1h[(16 + base + j) * 16 + nrow];
  }

  const int gwave = blockIdx.x * 4 + w;
  int cur = 0;

  // Prologue: stage iteration 0's member tile.
  {
    const int b0   = gwave;
    const int urow = members[b0 * MM + nrow];
    stage_member_row(U, urow, &memf[w][0][nrow][0], halfi);
  }

  for (int iter = 0; iter < K1_ITERS; ++iter) {
    const int b    = iter * K1_WAVES + gwave;   // uniform per wave
    const int item = item_in[b];
    const int grp  = group_in[b];
    const int len  = lens[b];

    const float itd = I[(size_t)item * DD + lane];   // lane owns dim d=lane
    const float gd  = G[(size_t)grp  * DD + lane];
    itf[w][lane] = itd;

    // Prefetch next tile into the other buffer, then wait for the current
    // buffer (in-order completion: cnt<=4 leaves only the new 4 in flight).
    if (iter + 1 < K1_ITERS) {
      const int bn    = (iter + 1) * K1_WAVES + gwave;
      const int urown = members[bn * MM + nrow];
      stage_member_row(U, urown, &memf[w][cur ^ 1][nrow][0], halfi);
      wait_async<4>();
    } else {
      wait_async<0>();
    }
    __syncthreads();   // uniform across block (all waves same trip count)

    // A fragments: tile0 = member cols 0..31, tile1 = item (rows identical)
    v16h fa0, fa1;
#pragma unroll
    for (int j = 0; j < 8; ++j) {
      fa0[j]     = (_Float16)memf[w][cur][nrow][base + j];
      fa0[8 + j] = (_Float16)memf[w][cur][nrow][16 + base + j];
      fa1[j]     = (_Float16)itf[w][base + j];
      fa1[8 + j] = (_Float16)itf[w][16 + base + j];
    }

    // ---- attention MLP stage 1: hidden(16x16) = [mem | it](16x64) @ Wa1 ----
    v8f acc = {0.f, 0.f, 0.f, 0.f, 0.f, 0.f, 0.f, 0.f};
    acc = __builtin_amdgcn_wmma_f32_16x16x32_f16(false, fa0, false, fwa0,
                                                 (short)0, acc, false, false);
    acc = __builtin_amdgcn_wmma_f32_16x16x32_f16(false, fa1, false, fwa1,
                                                 (short)0, acc, false, false);
    // stage 2: logit[m] = sum_n relu(h[m][n]+ba1[n]) * Wa2[n]   (+ba2: argmax-invariant)
    float att[8];
#pragma unroll
    for (int r = 0; r < 8; ++r) {
      float t = fmaxf(acc[r] + ba1s[nrow], 0.f) * wa2s[nrow];
      att[r] = half_sum16(t);        // logit for m = r + 8*halfi
    }
    // argmax over valid members (first-max tie break, like jnp.argmax)
    float bv = -3.0e38f; int bm = 0;
#pragma unroll
    for (int r = 0; r < 8; ++r) {
      const int m = r + 8 * halfi;
      const float v = (m < len) ? att[r] : NEGF;
      if (v > bv) { bv = v; bm = m; }
    }
    {
      const float ov = __shfl_xor(bv, 16, 32);
      const int   om = __shfl_xor(bm, 16, 32);
      if (ov > bv || (ov == bv && om < bm)) { bv = ov; bm = om; }
    }
    const int leader = bm;

    // ---- latent MLP: hidden = mem(16x32) @ Wu1, reuses fa0 ----
    v8f uacc = {0.f, 0.f, 0.f, 0.f, 0.f, 0.f, 0.f, 0.f};
    uacc = __builtin_amdgcn_wmma_f32_16x16x32_f16(false, fa0, false, fwu,
                                                  (short)0, uacc, false, false);
    float ul[8];
#pragma unroll
    for (int r = 0; r < 8; ++r) {
      float t = fmaxf(uacc[r] + bu1s[nrow], 0.f) * wu2s[nrow];
      ul[r] = half_sum16(t);         // u_logit for m = r + 8*halfi (+bu2 cancels in softmax)
    }
    // masked softmax over latent members (valid && m != leader); masked -> exactly 0
    bool vr[8]; float mx = -3.0e38f;
#pragma unroll
    for (int r = 0; r < 8; ++r) {
      const int m = r + 8 * halfi;
      vr[r] = (m < len) && (m != leader);
      if (vr[r]) mx = fmaxf(mx, ul[r]);
    }
    mx = fmaxf(mx, __shfl_xor(mx, 16, 32));
    float er[8]; float ssum = 0.f;
#pragma unroll
    for (int r = 0; r < 8; ++r) {
      er[r] = vr[r] ? __expf(ul[r] - mx) : 0.f;
      ssum += er[r];
    }
    ssum += __shfl_xor(ssum, 16, 32);
    const float inv = 1.f / ssum;
    float cr[8];
#pragma unroll
    for (int r = 0; r < 8; ++r) cr[r] = vr[r] ? (er[r] * inv + 1.f) : 0.f;

    // broadcast (w_m + 1)*mask coefficients to every lane (v_readlane)
    float coef[16];
#pragma unroll
    for (int m = 0; m < 16; ++m)
      coef[m] = bcast_lane(cr[m & 7], (m < 8) ? 0 : 16);

    // g_ilm[d] = (sum_m coef[m]*mem[m][d] + mem[leader][d]) / 2 ; lane owns d
    float gsum = 0.f;
#pragma unroll
    for (int m = 0; m < 16; ++m) gsum += coef[m] * memf[w][cur][m][lane];
    const float gil = 0.5f * (gsum + memf[w][cur][leader][lane]);
    gilm[(size_t)b * DD + lane] = gil;

    // ---- predict MLP: sigmoid(relu([g*i, g, i](96) @ Wp1 + bp1) @ Wp2 + bp2)
    const float gi = gd * itd;
    float outp = 0.f;
#pragma unroll
    for (int h = 0; h < 8; ++h) {
      const float p = gi  * wp1s[lane * 8 + h]
                    + gd  * wp1s[(32 + lane) * 8 + h]
                    + itd * wp1s[(64 + lane) * 8 + h];
      const float hv = wave_sum32(p);
      outp += fmaxf(hv + bp1s[h], 0.f) * wp2s[h];
    }
    const float val = outp + bp2v;
    if (lane == 0) sbuf[b] = 1.f / (1.f + __expf(-val));

    cur ^= 1;
    __syncthreads();   // keep per-wave LDS reuse ordered before next iteration
  }
}

// ---------------------------------------------------------------------------
// Kernel 2a: per-chunk (256-row) sums of g_ilm vectors
// ---------------------------------------------------------------------------
__global__ void __launch_bounds__(128)
chunk_sum_kernel(const float* __restrict__ gilm, float* __restrict__ chsum)
{
  __shared__ float part[4][32];
  const int tid = threadIdx.x, lane = tid & 31, w = tid >> 5;
  const int chunk = blockIdx.x;
  const size_t baseb = (size_t)chunk * CHUNK;
  float acc = 0.f;
  for (int r = w * 64; r < w * 64 + 64; ++r)
    acc += gilm[(baseb + r) * DD + lane];
  part[w][lane] = acc;
  __syncthreads();
  if (tid < 32)
    chsum[chunk * DD + tid] = part[0][tid] + part[1][tid] + part[2][tid] + part[3][tid];
}

// ---------------------------------------------------------------------------
// Kernel 2b: exclusive scan of the 256 chunk sums (per dim) — tiny, 1 wave
// ---------------------------------------------------------------------------
__global__ void __launch_bounds__(32)
scan_chunks_kernel(const float* __restrict__ chsum, float* __restrict__ chpre)
{
  const int d = threadIdx.x;
  float run = 0.f;
  for (int j = 0; j < NCHUNK; ++j) {
    const float t = chsum[j * DD + d];
    chpre[j * DD + d] = run;
    run += t;
  }
}

// ---------------------------------------------------------------------------
// Kernel 2c: running sum within chunk, R = |dot(cum, item)|, y = s + R
// ---------------------------------------------------------------------------
__global__ void __launch_bounds__(32)
finalize_kernel(const float* __restrict__ gilm, const float* __restrict__ chpre,
                const float* __restrict__ sbuf, const int* __restrict__ item_in,
                const float* __restrict__ I, float* __restrict__ out)
{
  const int j = blockIdx.x;   // chunk
  const int d = threadIdx.x;  // dim
  float run = chpre[j * DD + d];
  for (int r = 0; r < CHUNK; ++r) {
    const int b = j * CHUNK + r;
    run += gilm[(size_t)b * DD + d];
    const float itv = I[(size_t)item_in[b] * DD + d];
    const float dot = wave_sum32(run * itv);
    if (d == 0) out[b] = sbuf[b] + fabsf(dot);
  }
}

// ---------------------------------------------------------------------------
extern "C" void kernel_launch(void* const* d_in, const int* in_sizes, int n_in,
                              void* d_out, int out_size, void* d_ws, size_t ws_size,
                              hipStream_t stream)
{
  (void)in_sizes; (void)n_in; (void)out_size; (void)ws_size;
  const int*   group_in = (const int*)d_in[0];
  const int*   item_in  = (const int*)d_in[1];
  const int*   members  = (const int*)d_in[2];
  const int*   lens     = (const int*)d_in[3];
  const float* U   = (const float*)d_in[4];
  const float* I   = (const float*)d_in[5];
  const float* G   = (const float*)d_in[6];
  const float* Wa1 = (const float*)d_in[7];
  const float* ba1 = (const float*)d_in[8];
  const float* Wa2 = (const float*)d_in[9];
  // d_in[10] = ba2 (argmax-invariant), d_in[14] = bu2 (softmax-invariant)
  const float* Wu1 = (const float*)d_in[11];
  const float* bu1 = (const float*)d_in[12];
  const float* Wu2 = (const float*)d_in[13];
  const float* Wp1 = (const float*)d_in[15];
  const float* bp1 = (const float*)d_in[16];
  const float* Wp2 = (const float*)d_in[17];
  const float* bp2 = (const float*)d_in[18];

  float* gilm  = (float*)d_ws;                    // B*32 floats (8 MB)
  float* sbuf  = gilm + (size_t)BATCH * DD;       // B floats
  float* chsum = sbuf + BATCH;                    // NCHUNK*32
  float* chpre = chsum + NCHUNK * DD;             // NCHUNK*32

  gilm_kernel<<<K1_BLOCKS, K1_THREADS, 0, stream>>>(
      group_in, item_in, members, lens, U, I, G,
      Wa1, ba1, Wa2, Wu1, bu1, Wu2, Wp1, bp1, Wp2, bp2, gilm, sbuf);
  chunk_sum_kernel<<<NCHUNK, 128, 0, stream>>>(gilm, chsum);
  scan_chunks_kernel<<<1, 32, 0, stream>>>(chsum, chpre);
  finalize_kernel<<<NCHUNK, 32, 0, stream>>>(gilm, chpre, sbuf, item_in, I,
                                             (float*)d_out);
}
